// HAN_59854664237650
// MI455X (gfx1250) — matrix operand order repaired and checked by reading, not compiled, not run on hardware
//
#include <hip/hip_runtime.h>
#include <hip/hip_bf16.h>
#include <math.h>

typedef __attribute__((ext_vector_type(16))) _Float16 v16h;
typedef __attribute__((ext_vector_type(8)))  float    v8f;

#define IN_F   128
#define HID_F  64
#define HEADS  4
#define DH     16
#define OUT_F  32

// ---------- monotone float<->uint encoding for atomic max on signed floats ----------
__device__ __forceinline__ unsigned enc_f(float f) {
  unsigned u = __float_as_uint(f);
  return (u & 0x80000000u) ? ~u : (u | 0x80000000u);
}
__device__ __forceinline__ float dec_f(unsigned u) {
  u = (u & 0x80000000u) ? (u & 0x7FFFFFFFu) : ~u;
  return __uint_as_float(u);
}

__global__ void fill_u32_kernel(unsigned* __restrict__ p, unsigned v, int n) {
  int i = blockIdx.x * blockDim.x + threadIdx.x;
  if (i < n) p[i] = v;
}

// Build an f16 A-fragment (16x32 tile row slice) from 4 float4 loads.
__device__ __forceinline__ v16h load_a_frag_f32(const float* __restrict__ row,
                                                int kb, int khalf) {
  const float4* r4 = (const float4*)row;
  float4 p0 = r4[(kb + khalf) >> 2];
  float4 p1 = r4[((kb + khalf) >> 2) + 1];
  float4 p2 = r4[(kb + 16 + khalf) >> 2];
  float4 p3 = r4[((kb + 16 + khalf) >> 2) + 1];
  v16h a;
  a[0] = (_Float16)p0.x;  a[1] = (_Float16)p0.y;
  a[2] = (_Float16)p0.z;  a[3] = (_Float16)p0.w;
  a[4] = (_Float16)p1.x;  a[5] = (_Float16)p1.y;
  a[6] = (_Float16)p1.z;  a[7] = (_Float16)p1.w;
  a[8]  = (_Float16)p2.x; a[9]  = (_Float16)p2.y;
  a[10] = (_Float16)p2.z; a[11] = (_Float16)p2.w;
  a[12] = (_Float16)p3.x; a[13] = (_Float16)p3.y;
  a[14] = (_Float16)p3.z; a[15] = (_Float16)p3.w;
  return a;
}

// ---------- Phase 1: H = X @ W + b  via v_wmma_f32_16x16x32_f16 ----------
// B-fragments pre-swizzled in LDS: [kchunk][ntile][lane][16 halves] -> per-lane
// contiguous 32B, lane-stride 32B (conflict-free ds_load_b128 pairs).
__global__ __launch_bounds__(256)
void proj_wmma_kernel(const float* __restrict__ X, const float* __restrict__ W,
                      const float* __restrict__ bias, float* __restrict__ H, int n) {
  __shared__ __attribute__((aligned(32))) _Float16 sFrag[4 * 4 * 32 * 16]; // 16 KB
  __shared__ float sb[HID_F];
  int tid = threadIdx.x;
#pragma unroll
  for (int i = 0; i < 32; ++i) {               // 8192 halves / 256 threads
    int t    = i * 256 + tid;
    int idx  = t & 15;
    int ln   = (t >> 4) & 31;
    int nt   = (t >> 9) & 3;
    int kbc  = t >> 11;
    int k = kbc * 32 + ((ln >> 4) << 3) + (idx & 7) + ((idx >> 3) << 4); // B layout: table order
    int nn = nt * 16 + (ln & 15);
    sFrag[t] = (_Float16)W[k * HID_F + nn];
  }
  if (tid < HID_F) sb[tid] = bias[tid];
  __syncthreads();

  int wave = tid >> 5, lane = tid & 31;
  int rowbase = blockIdx.x * 128 + wave * 16;
  int lrow  = lane & 15;
  int khalf = (lane >> 4) << 3;
  int r = rowbase + lrow;
  if (r >= n) r = n - 1;     // clamp: EXEC stays all-ones for WMMA
  const float* xr = X + (size_t)r * IN_F;

  v8f acc[4];
#pragma unroll
  for (int t = 0; t < 4; ++t)
#pragma unroll
    for (int e = 0; e < 8; ++e) acc[t][e] = 0.f;

#pragma unroll
  for (int kbc = 0; kbc < 4; ++kbc) {
    v16h a = load_a_frag_f32(xr, kbc * 32, khalf);
    // preload all B fragments of this k-chunk -> one ds clause, one wait
    v16h bf[4];
#pragma unroll
    for (int nt = 0; nt < 4; ++nt)
      bf[nt] = *(const v16h*)&sFrag[((kbc * 4 + nt) * 32 + lane) * 16];
#pragma unroll
    for (int nt = 0; nt < 4; ++nt)
      acc[nt] = __builtin_amdgcn_wmma_f32_16x16x32_f16(
          false, a, false, bf[nt], (short)0, acc[nt], false, false);
  }

  int mbase = (lane < 16) ? 0 : 8;
#pragma unroll
  for (int nt = 0; nt < 4; ++nt) {
    int ncol = nt * 16 + lrow;
#pragma unroll
    for (int rr = 0; rr < 8; ++rr) {
      int m = rowbase + mbase + rr;
      if (m < n) H[(size_t)m * HID_F + ncol] = acc[nt][rr] + sb[ncol];
    }
  }
}

// ---------- Phase 2a: per-node attention terms ----------
__global__ void node_alpha_kernel(const float* __restrict__ H, const float* __restrict__ a,
                                  float* __restrict__ out, int n) {
  int i = blockIdx.x * blockDim.x + threadIdx.x;
  if (i >= n * HEADS) return;
  int node = i >> 2, h = i & 3;
  const float4* hp = (const float4*)(H + (size_t)node * HID_F + h * DH);
  const float4* ap = (const float4*)(a + h * DH);
  float s = 0.f;
#pragma unroll
  for (int d = 0; d < 4; ++d) {
    float4 hv = hp[d], av = ap[d];
    s += hv.x * av.x + hv.y * av.y + hv.z * av.z + hv.w * av.w;
  }
  out[i] = s;
}

__device__ __forceinline__ float leaky02(float x) { return (x >= 0.f) ? x : 0.2f * x; }

// ---------- Phase 2b: segment max ----------
__global__ void edge_max_kernel(const float* __restrict__ asrc, const float* __restrict__ adst,
                                const int* __restrict__ esrc, const int* __restrict__ edst,
                                unsigned* __restrict__ menc, int E) {
  int i = blockIdx.x * blockDim.x + threadIdx.x;
  if (i >= E * HEADS) return;
  int e = i >> 2, h = i & 3;
  int s = esrc[e], d = edst[e];
  float l = leaky02(asrc[s * HEADS + h] + adst[d * HEADS + h]);
  atomicMax(&menc[d * HEADS + h], enc_f(l));
}

// ---------- Phase 2c: segment exp-sum ----------
__global__ void edge_expsum_kernel(const float* __restrict__ asrc, const float* __restrict__ adst,
                                   const int* __restrict__ esrc, const int* __restrict__ edst,
                                   const unsigned* __restrict__ menc, float* __restrict__ z, int E) {
  int i = blockIdx.x * blockDim.x + threadIdx.x;
  if (i >= E * HEADS) return;
  int e = i >> 2, h = i & 3;
  int s = esrc[e], d = edst[e];
  float l = leaky02(asrc[s * HEADS + h] + adst[d * HEADS + h]);
  float m = dec_f(menc[d * HEADS + h]);
  if (!isfinite(m)) m = 0.f;
  atomicAdd(&z[d * HEADS + h], expf(l - m));
}

// ---------- Phase 2d: weighted message scatter-add ----------
__global__ void edge_msg_kernel(const float* __restrict__ Hsrc,
                                const float* __restrict__ asrc, const float* __restrict__ adst,
                                const int* __restrict__ esrc, const int* __restrict__ edst,
                                const unsigned* __restrict__ menc, const float* __restrict__ z,
                                float* __restrict__ out, int E) {
  int i = blockIdx.x * blockDim.x + threadIdx.x;
  if (i >= E * HEADS) return;
  int e = i >> 2, h = i & 3;
  int s = esrc[e], d = edst[e];
  float l = leaky02(asrc[s * HEADS + h] + adst[d * HEADS + h]);
  float m = dec_f(menc[d * HEADS + h]);
  if (!isfinite(m)) m = 0.f;
  float w = expf(l - m) / (z[d * HEADS + h] + 1e-16f);
  const float4* hs = (const float4*)(Hsrc + (size_t)s * HID_F + h * DH);
  float* op = out + (size_t)d * HID_F + h * DH;
#pragma unroll
  for (int v = 0; v < 4; ++v) {
    float4 hv = hs[v];
    atomicAdd(&op[4 * v + 0], hv.x * w);
    atomicAdd(&op[4 * v + 1], hv.y * w);
    atomicAdd(&op[4 * v + 2], hv.z * w);
    atomicAdd(&op[4 * v + 3], hv.w * w);
  }
}

// ---------- Phase 3: S[j] += sum_n tanh( (relu(out)[n] @ Wk + bk)[j] )  via WMMA ----------
__global__ __launch_bounds__(256)
void tanh_colsum_wmma_kernel(const float* __restrict__ H, const float* __restrict__ Wk,
                             const float* __restrict__ bk, float* __restrict__ S, int n) {
  __shared__ __attribute__((aligned(32))) _Float16 sFrag[2 * 4 * 32 * 16]; // 8 KB
  __shared__ float sb[HID_F];
  __shared__ float sred[HID_F];
  int tid = threadIdx.x;
#pragma unroll
  for (int i = 0; i < 16; ++i) {               // 4096 halves / 256 threads
    int t    = i * 256 + tid;
    int idx  = t & 15;
    int ln   = (t >> 4) & 31;
    int nt   = (t >> 9) & 3;
    int kbc  = t >> 11;
    int k = kbc * 32 + ((ln >> 4) << 3) + (idx & 7) + ((idx >> 3) << 4);
    int nn = nt * 16 + (ln & 15);
    sFrag[t] = (_Float16)Wk[k * HID_F + nn];
  }
  if (tid < HID_F) { sb[tid] = bk[tid]; sred[tid] = 0.f; }
  __syncthreads();

  int wave = tid >> 5, lane = tid & 31;
  int rowbase = blockIdx.x * 128 + wave * 16;
  int lrow  = lane & 15;
  int khalf = (lane >> 4) << 3;
  int r = rowbase + lrow;
  if (r >= n) r = n - 1;
  const float* hr = H + (size_t)r * HID_F;

  v8f acc[4];
#pragma unroll
  for (int t = 0; t < 4; ++t)
#pragma unroll
    for (int e = 0; e < 8; ++e) acc[t][e] = 0.f;

#pragma unroll
  for (int kbc = 0; kbc < 2; ++kbc) {
    // fused ReLU on the A operand
    const float4* r4 = (const float4*)hr;
    float4 p0 = r4[(kbc * 32 + khalf) >> 2];
    float4 p1 = r4[((kbc * 32 + khalf) >> 2) + 1];
    float4 p2 = r4[(kbc * 32 + 16 + khalf) >> 2];
    float4 p3 = r4[((kbc * 32 + 16 + khalf) >> 2) + 1];
    v16h a;
    a[0] = (_Float16)fmaxf(p0.x, 0.f);  a[1] = (_Float16)fmaxf(p0.y, 0.f);
    a[2] = (_Float16)fmaxf(p0.z, 0.f);  a[3] = (_Float16)fmaxf(p0.w, 0.f);
    a[4] = (_Float16)fmaxf(p1.x, 0.f);  a[5] = (_Float16)fmaxf(p1.y, 0.f);
    a[6] = (_Float16)fmaxf(p1.z, 0.f);  a[7] = (_Float16)fmaxf(p1.w, 0.f);
    a[8]  = (_Float16)fmaxf(p2.x, 0.f); a[9]  = (_Float16)fmaxf(p2.y, 0.f);
    a[10] = (_Float16)fmaxf(p2.z, 0.f); a[11] = (_Float16)fmaxf(p2.w, 0.f);
    a[12] = (_Float16)fmaxf(p3.x, 0.f); a[13] = (_Float16)fmaxf(p3.y, 0.f);
    a[14] = (_Float16)fmaxf(p3.z, 0.f); a[15] = (_Float16)fmaxf(p3.w, 0.f);
    v16h bf[4];
#pragma unroll
    for (int nt = 0; nt < 4; ++nt)
      bf[nt] = *(const v16h*)&sFrag[((kbc * 4 + nt) * 32 + lane) * 16];
#pragma unroll
    for (int nt = 0; nt < 4; ++nt)
      acc[nt] = __builtin_amdgcn_wmma_f32_16x16x32_f16(
          false, a, false, bf[nt], (short)0, acc[nt], false, false);
  }

  int mbase = (lane < 16) ? 0 : 8;
#pragma unroll
  for (int nt = 0; nt < 4; ++nt) {
    int ncol = nt * 16 + lrow;
    float local = 0.f;
#pragma unroll
    for (int rr = 0; rr < 8; ++rr) {
      int m = rowbase + mbase + rr;
      if (m < n) local += tanhf(acc[nt][rr] + sb[ncol]);
    }
    atomicAdd(&sred[ncol], local);  // LDS ds_add_f32
  }
  __syncthreads();
  if (tid < HID_F) atomicAdd(&S[tid], sred[tid]);
}

// ---------- Phase 4: 2-way semantic softmax ----------
__global__ void score_kernel(const float* __restrict__ S, const float* __restrict__ q,
                             float* __restrict__ attn, int nspot) {
  if (threadIdx.x == 0) {
    float s0 = 0.f, s1 = 0.f;
    for (int j = 0; j < HID_F; ++j) { s0 += q[j] * S[j]; s1 += q[j] * S[HID_F + j]; }
    s0 /= (float)nspot; s1 /= (float)nspot;
    float mx = fmaxf(s0, s1);
    float e0 = expf(s0 - mx), e1 = expf(s1 - mx);
    attn[0] = e0 / (e0 + e1);
    attn[1] = e1 / (e0 + e1);
  }
}

// ---------- Phase 5: Y = (a0*relu(U) + a1*relu(V)) @ W_lin + b_lin  via WMMA ----------
__global__ __launch_bounds__(256)
void final_wmma_kernel(const float* __restrict__ U, const float* __restrict__ V,
                       const float* __restrict__ attn,
                       const float* __restrict__ Wl, const float* __restrict__ bl,
                       float* __restrict__ Y, int n) {
  __shared__ __attribute__((aligned(32))) _Float16 sFrag[2 * 2 * 32 * 16]; // 4 KB
  __shared__ float sb[OUT_F];
  int tid = threadIdx.x;
#pragma unroll
  for (int i = 0; i < 8; ++i) {                // 2048 halves / 256 threads
    int t    = i * 256 + tid;
    int idx  = t & 15;
    int ln   = (t >> 4) & 31;
    int nt   = (t >> 9) & 1;
    int kbc  = t >> 10;
    int k = kbc * 32 + ((ln >> 4) << 3) + (idx & 7) + ((idx >> 3) << 4);
    int nn = nt * 16 + (ln & 15);
    sFrag[t] = (_Float16)Wl[k * OUT_F + nn];
  }
  if (tid < OUT_F) sb[tid] = bl[tid];
  __syncthreads();
  float a0 = attn[0], a1 = attn[1];

  int wave = tid >> 5, lane = tid & 31;
  int rowbase = blockIdx.x * 128 + wave * 16;
  int lrow  = lane & 15;
  int khalf = (lane >> 4) << 3;
  int r = rowbase + lrow;
  if (r >= n) r = n - 1;
  const float4* ur = (const float4*)(U + (size_t)r * HID_F);
  const float4* vr = (const float4*)(V + (size_t)r * HID_F);

  v8f acc[2];
#pragma unroll
  for (int t = 0; t < 2; ++t)
#pragma unroll
    for (int e = 0; e < 8; ++e) acc[t][e] = 0.f;

#pragma unroll
  for (int kbc = 0; kbc < 2; ++kbc) {
    v16h a;
#pragma unroll
    for (int half = 0; half < 2; ++half) {
      int base4 = (kbc * 32 + half * 16 + khalf) >> 2;
#pragma unroll
      for (int p = 0; p < 2; ++p) {
        float4 uu = ur[base4 + p], vv = vr[base4 + p];
        int o = half * 8 + p * 4;
        a[o + 0] = (_Float16)(a0 * fmaxf(uu.x, 0.f) + a1 * fmaxf(vv.x, 0.f));
        a[o + 1] = (_Float16)(a0 * fmaxf(uu.y, 0.f) + a1 * fmaxf(vv.y, 0.f));
        a[o + 2] = (_Float16)(a0 * fmaxf(uu.z, 0.f) + a1 * fmaxf(vv.z, 0.f));
        a[o + 3] = (_Float16)(a0 * fmaxf(uu.w, 0.f) + a1 * fmaxf(vv.w, 0.f));
      }
    }
    v16h bf[2];
#pragma unroll
    for (int nt = 0; nt < 2; ++nt)
      bf[nt] = *(const v16h*)&sFrag[((kbc * 2 + nt) * 32 + lane) * 16];
#pragma unroll
    for (int nt = 0; nt < 2; ++nt)
      acc[nt] = __builtin_amdgcn_wmma_f32_16x16x32_f16(
          false, a, false, bf[nt], (short)0, acc[nt], false, false);
  }

  int mbase = (lane < 16) ? 0 : 8;
#pragma unroll
  for (int nt = 0; nt < 2; ++nt) {
    int ncol = nt * 16 + lrow;
#pragma unroll
    for (int rr = 0; rr < 8; ++rr) {
      int m = rowbase + mbase + rr;
      if (m < n) Y[(size_t)m * OUT_F + ncol] = acc[nt][rr] + sb[ncol];
    }
  }
}

extern "C" void kernel_launch(void* const* d_in, const int* in_sizes, int n_in,
                              void* d_out, int out_size, void* d_ws, size_t ws_size,
                              hipStream_t stream) {
  const float* x_user     = (const float*)d_in[0];
  const float* x_spot     = (const float*)d_in[1];
  const float* W_user     = (const float*)d_in[2];
  const float* b_user     = (const float*)d_in[3];
  const float* W_spot     = (const float*)d_in[4];
  const float* b_spot     = (const float*)d_in[5];
  const float* att_src_us = (const float*)d_in[6];
  const float* att_dst_us = (const float*)d_in[7];
  const float* att_src_ss = (const float*)d_in[8];
  const float* att_dst_ss = (const float*)d_in[9];
  const float* Wk         = (const float*)d_in[10];
  const float* bk         = (const float*)d_in[11];
  const float* q          = (const float*)d_in[12];
  const float* W_lin      = (const float*)d_in[13];
  const float* b_lin      = (const float*)d_in[14];
  const int*   esrc_us    = (const int*)d_in[15];
  const int*   edst_us    = (const int*)d_in[16];
  const int*   esrc_ss    = (const int*)d_in[17];
  const int*   edst_ss    = (const int*)d_in[18];

  const int NU = in_sizes[0] / IN_F;
  const int NS = in_sizes[1] / IN_F;
  const int E  = in_sizes[15];

  char* ws = (char*)d_ws;
  size_t off = 0;
  auto carve = [&](size_t bytes) -> void* {
    void* p = ws + off;
    off = (off + bytes + 255) & ~(size_t)255;
    return p;
  };
  float*    h_user  = (float*)carve((size_t)NU * HID_F * 4);
  float*    h_spot  = (float*)carve((size_t)NS * HID_F * 4);
  float*    asrc_us = (float*)carve((size_t)NU * HEADS * 4);
  float*    adst_us = (float*)carve((size_t)NS * HEADS * 4);
  float*    asrc_ss = (float*)carve((size_t)NS * HEADS * 4);
  float*    adst_ss = (float*)carve((size_t)NS * HEADS * 4);
  unsigned* m_us    = (unsigned*)carve((size_t)NS * HEADS * 4);
  unsigned* m_ss    = (unsigned*)carve((size_t)NS * HEADS * 4);
  float*    z_us    = (float*)carve((size_t)NS * HEADS * 4);
  float*    z_ss    = (float*)carve((size_t)NS * HEADS * 4);
  float*    out_us  = (float*)carve((size_t)NS * HID_F * 4);
  float*    out_ss  = (float*)carve((size_t)NS * HID_F * 4);
  float*    S       = (float*)carve(2 * HID_F * 4);
  float*    attn    = (float*)carve(2 * 4);

  const int nh_s = NS * HEADS;
  const int nh_u = NU * HEADS;
  const int nsh  = NS * HID_F;
  const unsigned ENC_NEG_INF = 0x007FFFFFu;  // enc(-inf)

  // init accumulators/state (re-done every call: deterministic, graph-safe)
  fill_u32_kernel<<<(nh_s + 255) / 256, 256, 0, stream>>>(m_us, ENC_NEG_INF, nh_s);
  fill_u32_kernel<<<(nh_s + 255) / 256, 256, 0, stream>>>(m_ss, ENC_NEG_INF, nh_s);
  fill_u32_kernel<<<(nh_s + 255) / 256, 256, 0, stream>>>((unsigned*)z_us, 0u, nh_s);
  fill_u32_kernel<<<(nh_s + 255) / 256, 256, 0, stream>>>((unsigned*)z_ss, 0u, nh_s);
  fill_u32_kernel<<<(nsh + 255) / 256, 256, 0, stream>>>((unsigned*)out_us, 0u, nsh);
  fill_u32_kernel<<<(nsh + 255) / 256, 256, 0, stream>>>((unsigned*)out_ss, 0u, nsh);
  fill_u32_kernel<<<1, 256, 0, stream>>>((unsigned*)S, 0u, 2 * HID_F);

  // node projections (WMMA)
  proj_wmma_kernel<<<(NU + 127) / 128, 256, 0, stream>>>(x_user, W_user, b_user, h_user, NU);
  proj_wmma_kernel<<<(NS + 127) / 128, 256, 0, stream>>>(x_spot, W_spot, b_spot, h_spot, NS);

  // per-node attention terms
  node_alpha_kernel<<<(nh_u + 255) / 256, 256, 0, stream>>>(h_user, att_src_us, asrc_us, NU);
  node_alpha_kernel<<<(nh_s + 255) / 256, 256, 0, stream>>>(h_spot, att_dst_us, adst_us, NS);
  node_alpha_kernel<<<(nh_s + 255) / 256, 256, 0, stream>>>(h_spot, att_src_ss, asrc_ss, NS);
  node_alpha_kernel<<<(nh_s + 255) / 256, 256, 0, stream>>>(h_spot, att_dst_ss, adst_ss, NS);

  // edge softmax + message passing (L2-resident gathers/atomics)
  int eg = (E * HEADS + 255) / 256;
  edge_max_kernel<<<eg, 256, 0, stream>>>(asrc_us, adst_us, esrc_us, edst_us, m_us, E);
  edge_max_kernel<<<eg, 256, 0, stream>>>(asrc_ss, adst_ss, esrc_ss, edst_ss, m_ss, E);
  edge_expsum_kernel<<<eg, 256, 0, stream>>>(asrc_us, adst_us, esrc_us, edst_us, m_us, z_us, E);
  edge_expsum_kernel<<<eg, 256, 0, stream>>>(asrc_ss, adst_ss, esrc_ss, edst_ss, m_ss, z_ss, E);
  edge_msg_kernel<<<eg, 256, 0, stream>>>(h_user, asrc_us, adst_us, esrc_us, edst_us, m_us, z_us, out_us, E);
  edge_msg_kernel<<<eg, 256, 0, stream>>>(h_spot, asrc_ss, adst_ss, esrc_ss, edst_ss, m_ss, z_ss, out_ss, E);

  // semantic attention (WMMA GEMM + tanh + column reduction); ReLU fused into loads
  tanh_colsum_wmma_kernel<<<(NS + 127) / 128, 256, 0, stream>>>(out_us, Wk, bk, S, NS);
  tanh_colsum_wmma_kernel<<<(NS + 127) / 128, 256, 0, stream>>>(out_ss, Wk, bk, S + HID_F, NS);
  score_kernel<<<1, 32, 0, stream>>>(S, q, attn, NS);

  // fused blend + ReLU + output projection (WMMA)
  final_wmma_kernel<<<(NS + 127) / 128, 256, 0, stream>>>(out_us, out_ss, attn, W_lin, b_lin,
                                                          (float*)d_out, NS);
}
